// Siglip2Attention_67456756351104
// MI455X (gfx1250) — matrix operand order repaired and checked by reading, not compiled, not run on hardware
//
#include <hip/hip_runtime.h>

typedef __attribute__((ext_vector_type(16))) _Float16 v16h;
typedef __attribute__((ext_vector_type(8)))  _Float16 v8h;
typedef __attribute__((ext_vector_type(8)))  float    v8f;

#define S_LEN 16384
#define EMB   1152
#define NH    16
#define HD    72
#define BL    1024

__device__ __forceinline__ v16h cat8(v8h lo, v8h hi) {
  return __builtin_shufflevector(lo, hi, 0,1,2,3,4,5,6,7,8,9,10,11,12,13,14,15);
}
__device__ __forceinline__ v8h zero8h() {
  v8h z;
#pragma unroll
  for (int i = 0; i < 8; ++i) z[i] = (_Float16)0;
  return z;
}
__device__ __forceinline__ v8f zero8f() {
  v8f z;
#pragma unroll
  for (int i = 0; i < 8; ++i) z[i] = 0.0f;
  return z;
}
__device__ __forceinline__ v8f wmma_f16(v16h a, v16h b, v8f c) {
  return __builtin_amdgcn_wmma_f32_16x16x32_f16(false, a, false, b, (short)0, c,
                                                false, false);
}

// CDNA5 async global->LDS copy (ASYNCcnt path). VDST = VGPR with LDS byte
// address (low 32 bits of a generic pointer to __shared__), VADDR = 64-bit
// global address.  CDNA5 ISA.
__device__ __forceinline__ void async_b128(const _Float16* gptr,
                                           const _Float16* lptr) {
  uint32_t lds_off = (uint32_t)(uintptr_t)lptr;
  asm volatile("global_load_async_to_lds_b128 %0, %1, off"
               :
               : "v"(lds_off), "v"(gptr)
               : "memory");
}
__device__ __forceinline__ void wait_async0() {
  asm volatile("s_wait_asynccnt 0x0" ::: "memory");
}

// ---------------------------------------------------------------- convert
__global__ __launch_bounds__(256) void cvt_kernel(const float* __restrict__ in,
                                                  _Float16* __restrict__ out,
                                                  int n) {
  int i = blockIdx.x * 256 + threadIdx.x;
  if (i < n) out[i] = (_Float16)in[i];
}

// ---------------------------------------------------------------- GEMM
// C[M,N] = A[M,K] @ B[K,N] + bias.  BM=128 BN=128 BK=32, 8 waves, each wave
// 32x64 = 2x4 WMMA tiles (8 WMMAs per K-step, 6 fragment loads).
// Double-buffered LDS; A tile staged with global_load_async_to_lds_b128.
template <bool OUT_F32>
__global__ __launch_bounds__(256) void gemm_kernel(
    const _Float16* __restrict__ A, const _Float16* __restrict__ B,
    const float* __restrict__ bias, _Float16* __restrict__ Ch,
    float* __restrict__ Cf, int M, int N, int K) {
  __shared__ __align__(16) _Float16 Asub[2][128][40];
  __shared__ __align__(16) _Float16 Bt[2][128][40];  // transposed: [n][k]

  const int tid = threadIdx.x;
  const int lane = tid & 31;
  const int wave = tid >> 5;
  const int l16 = lane & 15;
  const int hs = lane >> 4;  // lane-half select
  const int m0 = blockIdx.y * 128;
  const int n0 = blockIdx.x * 128;
  const int wm = (wave & 3) * 32;
  const int wn = (wave >> 2) * 64;

  auto stage = [&](int buf, int kb) {
    {  // A tile 128x32: async DMA, 2 x b128 per thread
      int row = tid >> 1;
      int off = (tid & 1) * 16;
      const _Float16* src = A + (size_t)(m0 + row) * K + kb + off;
      async_b128(src, &Asub[buf][row][off]);
      async_b128(src + 8, &Asub[buf][row][off + 8]);
    }
#pragma unroll
    for (int it = 0; it < 2; ++it) {  // B tile 32x128 -> transpose to Bt[n][k]
      int c = tid + it * 256;
      int kr = c >> 4;
      int nc = (c & 15) * 8;
      v8h wv =
          *reinterpret_cast<const v8h*>(B + (size_t)(kb + kr) * N + n0 + nc);
#pragma unroll
      for (int j = 0; j < 8; ++j) Bt[buf][nc + j][kr] = wv[j];
    }
  };

  v8f acc[2][4];
#pragma unroll
  for (int i = 0; i < 2; ++i)
#pragma unroll
    for (int j = 0; j < 4; ++j) acc[i][j] = zero8f();

  stage(0, 0);
  wait_async0();
  __syncthreads();

  const int nK = K / 32;
  for (int kbi = 0; kbi < nK; ++kbi) {
    int buf = kbi & 1;
    if (kbi + 1 < nK) stage(buf ^ 1, (kbi + 1) * 32);  // overlap with WMMAs

    v16h af[2], bf[4];
#pragma unroll
    for (int i = 0; i < 2; ++i) {
      // A layout: lanes 0-15 K 0-7/16-23, lanes 16-31 K 8-15/24-31
      const _Float16* ar = &Asub[buf][wm + i * 16 + l16][0];
      int off = hs * 8;
      v8h lo = *reinterpret_cast<const v8h*>(ar + off);
      v8h hi = *reinterpret_cast<const v8h*>(ar + off + 16);
      af[i] = cat8(lo, hi);
    }
#pragma unroll
    for (int j = 0; j < 4; ++j) {
      // B layout: lanes 0-15 K 0-7/16-23, lanes 16-31 K 8-15/24-31
      const _Float16* br = &Bt[buf][wn + j * 16 + l16][0];
      int kof = hs * 8;
      v8h lo = *reinterpret_cast<const v8h*>(br + kof);
      v8h hi = *reinterpret_cast<const v8h*>(br + kof + 16);
      bf[j] = cat8(lo, hi);
    }
#pragma unroll
    for (int i = 0; i < 2; ++i)
#pragma unroll
      for (int j = 0; j < 4; ++j) acc[i][j] = wmma_f16(af[i], bf[j], acc[i][j]);

    wait_async0();
    __syncthreads();
  }

#pragma unroll
  for (int i = 0; i < 2; ++i)
#pragma unroll
    for (int j = 0; j < 4; ++j) {
      int gc = n0 + wn + j * 16 + l16;
      float bv = bias[gc];
#pragma unroll
      for (int r = 0; r < 8; ++r) {
        int gr = m0 + wm + i * 16 + r + hs * 8;
        float val = acc[i][j][r] + bv;
        if constexpr (OUT_F32)
          Cf[(size_t)gr * N + gc] = val;
        else
          Ch[(size_t)gr * N + gc] = (_Float16)val;
      }
    }
}

// ---------------------------------------------------------------- RoPE
// in-place on q16/k16: each thread owns pair (d, d+36) of one head.
__global__ __launch_bounds__(256) void rope_kernel(
    _Float16* __restrict__ q, _Float16* __restrict__ k,
    const float* __restrict__ cosb, const float* __restrict__ sinb) {
  int idx = blockIdx.x * 256 + threadIdx.x;
  const int total = S_LEN * NH * (HD / 2);
  if (idx >= total) return;
  int s = idx / (NH * (HD / 2));
  int p = idx % (NH * (HD / 2));
  int h = p / (HD / 2);
  int d = p % (HD / 2);
  size_t base = (size_t)s * EMB + h * HD;
  float c1 = cosb[s * HD + d], s1 = sinb[s * HD + d];
  float c2 = cosb[s * HD + d + HD / 2], s2 = sinb[s * HD + d + HD / 2];
  {
    float x1 = (float)q[base + d], x2 = (float)q[base + d + HD / 2];
    q[base + d] = (_Float16)(x1 * c1 - x2 * s1);
    q[base + d + HD / 2] = (_Float16)(x2 * c2 + x1 * s2);
  }
  {
    float x1 = (float)k[base + d], x2 = (float)k[base + d + HD / 2];
    k[base + d] = (_Float16)(x1 * c1 - x2 * s1);
    k[base + d + HD / 2] = (_Float16)(x2 * c2 + x1 * s2);
  }
}

// ---------------------------------------------------------------- attention
// One workgroup (4 waves) = 64 query rows of one (block b, head h).
// Streaming (flash) softmax over 16 kv-steps of 64 keys.
// K tile staged via async-to-LDS; V staged transposed via ds stores.
__global__ __launch_bounds__(128) void attn_kernel(
    const _Float16* __restrict__ q, const _Float16* __restrict__ k,
    const _Float16* __restrict__ v, _Float16* __restrict__ ctx) {
  __shared__ __align__(16) _Float16 Kt[64][104];    // [key][d padded to 96]
  __shared__ __align__(16) _Float16 Vt[80][72];     // transposed [d pad 80][key]
  __shared__ __align__(16) _Float16 Pb[4][16][72];  // per-wave P tile

  const int tid = threadIdx.x;
  const int lane = tid & 31;
  const int wave = tid >> 5;
  const int l16 = lane & 15;
  const int hs = lane >> 4;

  const int bid = blockIdx.x;
  const int qc = bid & 15;
  const int h = (bid >> 4) & 15;
  const int b = bid >> 8;
  const int hcol = h * HD;

  // zero the pad regions once (never re-written in the loop)
  for (int i = tid; i < 64 * 24; i += 128) Kt[i / 24][72 + (i % 24)] = (_Float16)0;
  for (int i = tid; i < 8 * 64; i += 128) Vt[72 + (i >> 6)][i & 63] = (_Float16)0;

  // Q fragments (head-dim 72 padded to 96 -> 3 K-steps of 32), kept in VGPRs.
  const int qrow = b * BL + qc * 64 + wave * 16 + l16;
  const _Float16* qp = q + (size_t)qrow * EMB + hcol;
  v16h qa[3];
#pragma unroll
  for (int s = 0; s < 2; ++s) {
    int off = s * 32 + hs * 8;
    v8h lo = *reinterpret_cast<const v8h*>(qp + off);
    v8h hi = *reinterpret_cast<const v8h*>(qp + off + 16);
    qa[s] = cat8(lo, hi);
  }
  {
    v8h z = zero8h();
    v8h lo = hs ? z : *reinterpret_cast<const v8h*>(qp + 64);
    qa[2] = cat8(lo, z);
  }

  float mrow[8], lrow[8];
  v8f oacc[5];
#pragma unroll
  for (int r = 0; r < 8; ++r) { mrow[r] = -1e30f; lrow[r] = 0.0f; }
#pragma unroll
  for (int t = 0; t < 5; ++t) oacc[t] = zero8f();

  const float scale = 0.11785113019775793f;  // 72^-0.5

  for (int step = 0; step < 16; ++step) {
    __syncthreads();
    const int krow0 = b * BL + step * 64;
    // K tile (64 keys x 72 d) async to LDS; V tile transposed manually.
    for (int idx = tid; idx < 64 * 9; idx += 128) {
      int key = idx / 9, c = idx % 9;
      const _Float16* kp = k + (size_t)(krow0 + key) * EMB + hcol + c * 8;
      async_b128(kp, &Kt[key][c * 8]);
      v8h vv = *reinterpret_cast<const v8h*>(
          v + (size_t)(krow0 + key) * EMB + hcol + c * 8);
#pragma unroll
      for (int j = 0; j < 8; ++j) Vt[c * 8 + j][key] = vv[j];
    }
    wait_async0();
    __syncthreads();

    // scores: 16 q-rows x 64 keys per wave
    v8f sacc[4];
#pragma unroll
    for (int nt = 0; nt < 4; ++nt) sacc[nt] = zero8f();
#pragma unroll
    for (int nt = 0; nt < 4; ++nt) {
      const _Float16* kr = &Kt[nt * 16 + l16][0];
#pragma unroll
      for (int ks = 0; ks < 3; ++ks) {
        int kof = ks * 32 + hs * 8;
        v8h lo = *reinterpret_cast<const v8h*>(kr + kof);
        v8h hi = *reinterpret_cast<const v8h*>(kr + kof + 16);
        sacc[nt] = wmma_f16(qa[ks], cat8(lo, hi), sacc[nt]);
      }
    }

    // online softmax per row (C layout: lanes 0-15=cols, elem r = row r/r+8)
#pragma unroll
    for (int r = 0; r < 8; ++r) {
      float s0 = sacc[0][r] * scale;
      float s1 = sacc[1][r] * scale;
      float s2 = sacc[2][r] * scale;
      float s3 = sacc[3][r] * scale;
      float mx = fmaxf(fmaxf(s0, s1), fmaxf(s2, s3));
#pragma unroll
      for (int o = 8; o >= 1; o >>= 1) mx = fmaxf(mx, __shfl_xor(mx, o, 32));
      float mnew = fmaxf(mrow[r], mx);
      float alpha = __expf(mrow[r] - mnew);
      mrow[r] = mnew;
      float p0 = __expf(s0 - mnew);
      float p1 = __expf(s1 - mnew);
      float p2 = __expf(s2 - mnew);
      float p3 = __expf(s3 - mnew);
      int prow = r + hs * 8;
      Pb[wave][prow][0 + l16] = (_Float16)p0;
      Pb[wave][prow][16 + l16] = (_Float16)p1;
      Pb[wave][prow][32 + l16] = (_Float16)p2;
      Pb[wave][prow][48 + l16] = (_Float16)p3;
      float ps = p0 + p1 + p2 + p3;
#pragma unroll
      for (int o = 8; o >= 1; o >>= 1) ps += __shfl_xor(ps, o, 32);
      lrow[r] = lrow[r] * alpha + ps;
#pragma unroll
      for (int t = 0; t < 5; ++t) oacc[t][r] *= alpha;
    }

    // out += P(16x64) @ V(64x72pad80); P re-read from LDS in A layout
#pragma unroll
    for (int k2 = 0; k2 < 2; ++k2) {
      const _Float16* pr = &Pb[wave][l16][0];
      int off = k2 * 32 + hs * 8;
      v8h plo = *reinterpret_cast<const v8h*>(pr + off);
      v8h phi = *reinterpret_cast<const v8h*>(pr + off + 16);
      v16h pf = cat8(plo, phi);
      int kof = k2 * 32 + hs * 8;
#pragma unroll
      for (int dt = 0; dt < 5; ++dt) {
        const _Float16* vr = &Vt[dt * 16 + l16][0];
        v8h blo = *reinterpret_cast<const v8h*>(vr + kof);
        v8h bhi = *reinterpret_cast<const v8h*>(vr + kof + 16);
        oacc[dt] = wmma_f16(pf, cat8(blo, bhi), oacc[dt]);
      }
    }
  }

#pragma unroll
  for (int r = 0; r < 8; ++r) {
    int row = b * BL + qc * 64 + wave * 16 + r + hs * 8;
    float inv = 1.0f / lrow[r];
#pragma unroll
    for (int dt = 0; dt < 5; ++dt) {
      int d = dt * 16 + l16;
      if (d < HD)
        ctx[(size_t)row * EMB + hcol + d] = (_Float16)(oacc[dt][r] * inv);
    }
  }
}

// ---------------------------------------------------------------- launch
extern "C" void kernel_launch(void* const* d_in, const int* in_sizes, int n_in,
                              void* d_out, int out_size, void* d_ws,
                              size_t ws_size, hipStream_t stream) {
  const float* hidden = (const float*)d_in[0];
  // d_in[1] = cu_seqlens: uniform blocks of L, folded into indexing
  const float* cosb = (const float*)d_in[2];
  const float* sinb = (const float*)d_in[3];
  const float* Wq = (const float*)d_in[4];
  const float* bq = (const float*)d_in[5];
  const float* Wk = (const float*)d_in[6];
  const float* bk = (const float*)d_in[7];
  const float* Wv = (const float*)d_in[8];
  const float* bv = (const float*)d_in[9];
  const float* Wo = (const float*)d_in[10];
  const float* bo = (const float*)d_in[11];
  float* out = (float*)d_out;

  const size_t he = (size_t)S_LEN * EMB;  // 18,874,368
  const size_t ww = (size_t)EMB * EMB;    // 1,327,104
  char* w = (char*)d_ws;
  _Float16* h16 = (_Float16*)w;  w += he * 2;
  _Float16* q16 = (_Float16*)w;  w += he * 2;
  _Float16* k16 = (_Float16*)w;  w += he * 2;
  _Float16* v16 = (_Float16*)w;  w += he * 2;
  _Float16* c16 = (_Float16*)w;  w += he * 2;
  _Float16* wq16 = (_Float16*)w; w += ww * 2;
  _Float16* wk16 = (_Float16*)w; w += ww * 2;
  _Float16* wv16 = (_Float16*)w; w += ww * 2;
  _Float16* wo16 = (_Float16*)w; w += ww * 2;

  cvt_kernel<<<(int)((he + 255) / 256), 256, 0, stream>>>(hidden, h16, (int)he);
  cvt_kernel<<<(int)((ww + 255) / 256), 256, 0, stream>>>(Wq, wq16, (int)ww);
  cvt_kernel<<<(int)((ww + 255) / 256), 256, 0, stream>>>(Wk, wk16, (int)ww);
  cvt_kernel<<<(int)((ww + 255) / 256), 256, 0, stream>>>(Wv, wv16, (int)ww);
  cvt_kernel<<<(int)((ww + 255) / 256), 256, 0, stream>>>(Wo, wo16, (int)ww);

  dim3 gg(EMB / 128, S_LEN / 128);  // (9, 128)
  gemm_kernel<false><<<gg, 256, 0, stream>>>(h16, wq16, bq, q16, nullptr,
                                             S_LEN, EMB, EMB);
  gemm_kernel<false><<<gg, 256, 0, stream>>>(h16, wk16, bk, k16, nullptr,
                                             S_LEN, EMB, EMB);
  gemm_kernel<false><<<gg, 256, 0, stream>>>(h16, wv16, bv, v16, nullptr,
                                             S_LEN, EMB, EMB);

  const int npairs = S_LEN * NH * (HD / 2);
  rope_kernel<<<(npairs + 255) / 256, 256, 0, stream>>>(q16, k16, cosb, sinb);

  attn_kernel<<<16 * 16 * 16, 128, 0, stream>>>(q16, k16, v16, c16);

  gemm_kernel<true><<<gg, 256, 0, stream>>>(c16, wo16, bo, nullptr, out,
                                            S_LEN, EMB, EMB);
}
